// Net_wCB_43834436223508
// MI455X (gfx1250) — matrix operand order repaired and checked by reading, not compiled, not run on hardware
//
#include <hip/hip_runtime.h>
#include <hip/hip_bf16.h>

// ---------------------------------------------------------------------------
// GCN (3 layers) + normalized similarity for MI455X (gfx1250, wave32).
// All GEMMs: v_wmma_f32_16x16x32_bf16, double-buffered LDS, async-to-LDS
// staging (ASYNCcnt) where the copy is transpose-free.
// Sizes: B=64, N=1024, F=128 — every GEMM dim is a multiple of the tile.
// ---------------------------------------------------------------------------

#define BN 64
#define NN 1024
#define FF 128

typedef __attribute__((ext_vector_type(8)))  __bf16 v8bf;
typedef __attribute__((ext_vector_type(16))) __bf16 v16bf;
typedef __attribute__((ext_vector_type(8)))  float  v8f;
typedef int v4i __attribute__((vector_size(16)));   // 128-bit payload type

#define AS1 __attribute__((address_space(1)))
#define AS3 __attribute__((address_space(3)))

__device__ __forceinline__ unsigned short f2bf(float f) {
  unsigned u = __float_as_uint(f);
  u += 0x7FFFu + ((u >> 16) & 1u);      // round-to-nearest-even
  return (unsigned short)(u >> 16);
}
__device__ __forceinline__ float bf2f(unsigned short b) {
  return __uint_as_float(((unsigned)b) << 16);
}

__device__ __forceinline__ void st_out(float* p, float v)          { *p = v; }
__device__ __forceinline__ void st_out(unsigned short* p, float v) { *p = f2bf(v); }

// --- 16B global -> LDS copy via CDNA5 async path (ASYNCcnt) ----------------
__device__ __forceinline__ void cp16_async(const unsigned short* g, unsigned short* l) {
#if __has_builtin(__builtin_amdgcn_global_load_async_to_lds_b128)
  __builtin_amdgcn_global_load_async_to_lds_b128(
      (AS1 v4i*)(uintptr_t)g,                    // global src (128-bit elem)
      (AS3 v4i*)(unsigned int)(uintptr_t)l,      // LDS dst (low 32 bits = offset)
      0, 0);
#else
  asm volatile("global_load_async_to_lds_b128 %0, %1, off"
               :: "v"((unsigned int)(uintptr_t)l),          // VDST = LDS addr
                  "v"((unsigned long long)(uintptr_t)g)     // VADDR = 64b global
               : "memory");
#endif
}
__device__ __forceinline__ void wait_async0() {
#if __has_builtin(__builtin_amdgcn_s_wait_asynccnt)
  __builtin_amdgcn_s_wait_asynccnt(0);
#else
  asm volatile("s_wait_asynccnt 0x0" ::: "memory");
#endif
}

// ---------------------------------------------------------------------------
// Batched bf16 GEMM: C[z] = op(A[z] @ B[z]); 128x128 block tile, 256 threads
// = 8 wave32 in a 2x4 grid; each wave: 4x2 tiles of 16x16, K-step 32.
// TRANSB: B(k,n) = Bg[n*ldb + k] (register-staged transpose).
// Grid: (N/128, M/128, batch).
// ---------------------------------------------------------------------------
template <bool TRANSB, bool RELU, typename TOUT>
__global__ __launch_bounds__(256)
void gemm_bf16_wmma(const unsigned short* __restrict__ Ag, long sA, int lda,
                    const unsigned short* __restrict__ Bg, long sB, int ldb,
                    TOUT* __restrict__ Cg, long sC, int ldc, int K)
{
  __shared__ __align__(16) unsigned short As[2][128][40];   // +8 pad, 80B rows
  __shared__ __align__(16) unsigned short Bs[2][32][136];   // +8 pad, 272B rows

  const int t    = threadIdx.x;
  const int lane = t & 31;
  const int wave = t >> 5;
  const int wm   = (wave & 1) * 64;
  const int wn   = (wave >> 1) * 32;
  const int m0   = blockIdx.y * 128;
  const int n0   = blockIdx.x * 128;

  const unsigned short* Ab = Ag + (long)blockIdx.z * sA;
  const unsigned short* Bb = Bg + (long)blockIdx.z * sB;
  TOUT*                 Cb = Cg + (long)blockIdx.z * sC;

  // ---- staging: pure bf16 copies; async where transpose-free --------------
  auto stageA = [&](int k0, int buf) {
#pragma unroll
    for (int i = 0; i < 2; ++i) {            // 512 x 16B chunks, 2/thread
      int c = t + i * 256;
      int r = c >> 2;                        // 4 chunks per 64B row
      int e = (c & 3) * 8;                   // element offset (16B steps)
      cp16_async(&Ab[(long)(m0 + r) * lda + k0 + e], &As[buf][r][e]);
    }
  };
  auto stageB = [&](int k0, int buf) {
    if (!TRANSB) {
#pragma unroll
      for (int i = 0; i < 2; ++i) {          // 32 rows x 16 chunks
        int c = t + i * 256;
        int r = c >> 4;
        int e = (c & 15) * 8;
        cp16_async(&Bb[(long)(k0 + r) * ldb + n0 + e], &Bs[buf][r][e]);
      }
    } else {
      ushort4 v[4];
#pragma unroll
      for (int i = 0; i < 4; ++i) {          // phase 1: issue all loads
        int idx = (t + i * 256) << 2;
        int n = idx >> 5;
        int k = idx & 31;
        v[i] = *(const ushort4*)&Bb[(long)(n0 + n) * ldb + k0 + k];
      }
#pragma unroll
      for (int i = 0; i < 4; ++i) {          // phase 2: transposed LDS stores
        int idx = (t + i * 256) << 2;
        int n = idx >> 5;
        int k = idx & 31;
        Bs[buf][k + 0][n] = v[i].x; Bs[buf][k + 1][n] = v[i].y;
        Bs[buf][k + 2][n] = v[i].z; Bs[buf][k + 3][n] = v[i].w;
      }
    }
  };

  v8f acc[4][2];
#pragma unroll
  for (int i = 0; i < 4; ++i)
#pragma unroll
    for (int j = 0; j < 2; ++j)
#pragma unroll
      for (int e = 0; e < 8; ++e) acc[i][j][e] = 0.f;

  const int nt = K >> 5;
  stageA(0, 0);
  stageB(0, 0);

  const int kb   = (lane >> 4) * 8;
  const int lrow = lane & 15;
  union V16 { v16bf v; v8bf h[2]; };

  int cur = 0;
  for (int it = 0; it < nt; ++it) {
    wait_async0();          // this wave's async copies for tile `it` done
    __syncthreads();        // all waves' copies done (dscnt auto-waited too)

    if (it + 1 < nt) {      // overlap next tile's copies with WMMA
      stageA((it + 1) << 5, cur ^ 1);
      stageB((it + 1) << 5, cur ^ 1);
    }

    // Hoist ALL fragment loads (4A + 2B, distinct regs) so the ds_load_b128s
    // pipeline and the 8 WMMAs can issue back-to-back (§7.5 XDL issue).
    V16 bfr[2];
#pragma unroll
    for (int nt2 = 0; nt2 < 2; ++nt2) {      // B 32x16: lane = col N, K = kb..kb+7, kb+16..kb+23
#pragma unroll
      for (int s = 0; s < 8; ++s) {
        bfr[nt2].h[0][s] = __builtin_bit_cast(__bf16, Bs[cur][kb + s][wn + nt2 * 16 + lrow]);
        bfr[nt2].h[1][s] = __builtin_bit_cast(__bf16, Bs[cur][kb + 16 + s][wn + nt2 * 16 + lrow]);
      }
    }
    V16 afr[4];
#pragma unroll
    for (int mt = 0; mt < 4; ++mt) {
      int row = wm + mt * 16 + lrow;
      afr[mt].h[0] = *(const v8bf*)&As[cur][row][kb];        // K = kb..kb+7
      afr[mt].h[1] = *(const v8bf*)&As[cur][row][kb + 16];   // K = kb+16..kb+23
    }
#pragma unroll
    for (int mt = 0; mt < 4; ++mt)
#pragma unroll
      for (int nt2 = 0; nt2 < 2; ++nt2)
        acc[mt][nt2] = __builtin_amdgcn_wmma_f32_16x16x32_bf16(
            false, afr[mt].v, false, bfr[nt2].v, (short)0, acc[mt][nt2], false, false);

    cur ^= 1;
  }

  // ---- epilogue: D layout VGPR r -> (M = r + 8*(lane>=16), N = lane&15) ---
  const int mo = (lane >> 4) * 8;
  const int nc = lane & 15;
#pragma unroll
  for (int mt = 0; mt < 4; ++mt)
#pragma unroll
    for (int nt2 = 0; nt2 < 2; ++nt2)
#pragma unroll
      for (int r = 0; r < 8; ++r) {
        float v = acc[mt][nt2][r];
        if (RELU) v = fmaxf(v, 0.f);
        int row = m0 + wm + mt * 16 + mo + r;
        int col = n0 + wn + nt2 * 16 + nc;
        st_out(&Cb[(long)row * ldc + col], v);
      }
}

// ---------------------------------------------------------------------------
// fp32 -> bf16 bulk convert (4 elems/thread).
// ---------------------------------------------------------------------------
__global__ __launch_bounds__(256)
void f32_to_bf16(const float* __restrict__ src, unsigned short* __restrict__ dst, long n) {
  long i = ((long)blockIdx.x * 256 + threadIdx.x) * 4;
  if (i + 3 < n) {
    float4 v = *(const float4*)&src[i];
    ushort4 o; o.x = f2bf(v.x); o.y = f2bf(v.y); o.z = f2bf(v.z); o.w = f2bf(v.w);
    *(ushort4*)&dst[i] = o;
  }
}

// ---------------------------------------------------------------------------
// Row normalization over concat(h0, x3) [256 feats], ddof=1 std.
// Writes Zl = theta * z (bf16) and Zn = z (bf16). One block per (b,n) row.
// ---------------------------------------------------------------------------
__global__ __launch_bounds__(256)
void rownorm_kernel(const float* __restrict__ h0,
                    const unsigned short* __restrict__ x3,
                    const float* __restrict__ theta,
                    unsigned short* __restrict__ Zl,
                    unsigned short* __restrict__ Zn)
{
  __shared__ float red0[256];
  __shared__ float red1[256];
  const long row = blockIdx.x;
  const int  f   = threadIdx.x;
  float v = (f < FF) ? h0[row * FF + f] : bf2f(x3[row * FF + (f - FF)]);
  red0[f] = v;
  red1[f] = v * v;
  __syncthreads();
  for (int s = 128; s > 0; s >>= 1) {
    if (f < s) { red0[f] += red0[f + s]; red1[f] += red1[f + s]; }
    __syncthreads();
  }
  float mean = red0[0] * (1.f / 256.f);
  float var  = (red1[0] - 256.f * mean * mean) * (1.f / 255.f);
  float z    = (v - mean) * rsqrtf(var);
  Zl[row * 256 + f] = f2bf(z * theta[f]);
  Zn[row * 256 + f] = f2bf(z);
}

// ---------------------------------------------------------------------------
extern "C" void kernel_launch(void* const* d_in, const int* in_sizes, int n_in,
                              void* d_out, int out_size, void* d_ws, size_t ws_size,
                              hipStream_t stream) {
  const float* h0    = (const float*)d_in[0];   // [B,N,F]
  const float* A     = (const float*)d_in[1];   // [N,N]
  const float* W0    = (const float*)d_in[2];   // [F,F]
  const float* W1    = (const float*)d_in[3];
  const float* W2    = (const float*)d_in[4];
  const float* theta = (const float*)d_in[5];   // [2F]
  float* out = (float*)d_out;                   // [B,N,N]

  // Workspace (bf16 elements). Zl aliases {h0bf, Y} which are dead by then.
  unsigned short* Abf  = (unsigned short*)d_ws;            // 1M   (2 MB)
  unsigned short* W0bf = Abf  + (size_t)NN * NN;
  unsigned short* W1bf = W0bf + (size_t)FF * FF;
  unsigned short* W2bf = W1bf + (size_t)FF * FF;
  unsigned short* h0bf = W2bf + (size_t)FF * FF;           // 8M  (16 MB)
  unsigned short* Y    = h0bf + (size_t)BN * NN * FF;      // 8M  (16 MB)
  unsigned short* X    = Y    + (size_t)BN * NN * FF;      // 8M  (16 MB)
  unsigned short* Zn   = X    + (size_t)BN * NN * FF;      // 16M (32 MB)
  unsigned short* Zl   = h0bf;                             // 16M (32 MB, reuse)

  const long sBNF = (long)NN * FF;
  const long sZ   = (long)NN * 2 * FF;
  const long sOut = (long)NN * NN;

  dim3 blk(256);
  dim3 g1(1, NN / 128, BN);
  dim3 g3(NN / 128, NN / 128, BN);

  // ---- one-time bf16 conversion of fp32 operands --------------------------
  f32_to_bf16<<<dim3((NN * NN) / 1024), blk, 0, stream>>>(A, Abf, (long)NN * NN);
  f32_to_bf16<<<dim3((FF * FF) / 1024), blk, 0, stream>>>(W0, W0bf, (long)FF * FF);
  f32_to_bf16<<<dim3((FF * FF) / 1024), blk, 0, stream>>>(W1, W1bf, (long)FF * FF);
  f32_to_bf16<<<dim3((FF * FF) / 1024), blk, 0, stream>>>(W2, W2bf, (long)FF * FF);
  f32_to_bf16<<<dim3((BN * NN * FF) / 1024), blk, 0, stream>>>(h0, h0bf, (long)BN * NN * FF);

  // ---- Layer 1: Y = h0 @ W0 ; X = relu(A @ Y) -----------------------------
  gemm_bf16_wmma<false, false, unsigned short>
      <<<g1, blk, 0, stream>>>(h0bf, sBNF, FF, W0bf, 0, FF, Y, sBNF, FF, FF);
  gemm_bf16_wmma<false, true, unsigned short>
      <<<g1, blk, 0, stream>>>(Abf, 0, NN, Y, sBNF, FF, X, sBNF, FF, NN);

  // ---- Layer 2 ------------------------------------------------------------
  gemm_bf16_wmma<false, false, unsigned short>
      <<<g1, blk, 0, stream>>>(X, sBNF, FF, W1bf, 0, FF, Y, sBNF, FF, FF);
  gemm_bf16_wmma<false, true, unsigned short>
      <<<g1, blk, 0, stream>>>(Abf, 0, NN, Y, sBNF, FF, X, sBNF, FF, NN);

  // ---- Layer 3 ------------------------------------------------------------
  gemm_bf16_wmma<false, false, unsigned short>
      <<<g1, blk, 0, stream>>>(X, sBNF, FF, W2bf, 0, FF, Y, sBNF, FF, FF);
  gemm_bf16_wmma<false, true, unsigned short>
      <<<g1, blk, 0, stream>>>(Abf, 0, NN, Y, sBNF, FF, X, sBNF, FF, NN);

  // ---- Normalize + theta fold (h0bf/Y dead now; Zl overwrites them) -------
  rownorm_kernel<<<dim3(BN * NN), blk, 0, stream>>>(h0, X, theta, Zl, Zn);

  // ---- sim = Zl @ Zn^T ----------------------------------------------------
  gemm_bf16_wmma<true, false, float>
      <<<g3, blk, 0, stream>>>(Zl, sZ, 2 * FF, Zn, sZ, 2 * FF,
                               out, sOut, NN, 2 * FF);
}
